// GATTransformerODWeek_9131100471794
// MI455X (gfx1250) — matrix operand order repaired and checked by reading, not compiled, not run on hardware
//
// GATTransformerODWeek forward for AMD MI455X (gfx1250, CDNA5, wave32).
//
// Design (compile-only; reasoned from MI455X specs):
//  * Model is tiny (~2.5 GFLOP, ~100 MB traffic) vs 23.3 TB/s HBM + WMMA peaks
//    -> latency bound. One 16x16 output tile per wave32; all dense projections
//    via v_wmma_f32_16x16x32_bf16 (HID=32 == exactly one WMMA K-step), f32 acc.
//  * B (weight) operand is a strided column gather -> staged as a K x 16 f32
//    panel in LDS by the Tensor Data Mover (tensor_load_to_lds + s_wait_tensorcnt),
//    inner loop reads LDS. A operand read with global_load_b128 (two contiguous
//    8-float runs per lane in the WMMA A layout).
//  * All dense call sites have M%16==0 && N%16==0 -> branch-free epilogue
//    (8 pointer-incremented stores); guards only on K=72 convs / N=500 edges.
//  * GATv2 segment softmax: order-preserving uint atomicMax + f32 atomicAdd
//    (resident in 192MB L2).  Final OD einsum: one WMMA per 16x16 tile (K=32).
#include <hip/hip_runtime.h>
#include <hip/hip_bf16.h>
#include <math.h>
#include <cstdint>

#define CDIV(a,b) (((a)+(b)-1)/(b))

typedef __attribute__((ext_vector_type(16))) __bf16 v16bf;
typedef __attribute__((ext_vector_type(8)))  float  v8f;
typedef __attribute__((ext_vector_type(4)))  unsigned u32x4;
typedef __attribute__((ext_vector_type(8)))  int      i32x8;
typedef __attribute__((ext_vector_type(4)))  int      i32x4;

__device__ __forceinline__ __bf16 f2bf(float x) {
  unsigned u = __float_as_uint(x);
  unsigned r = (u + 0x7FFFu + ((u >> 16) & 1u)) >> 16;   // round-to-nearest-even
  unsigned short s = (unsigned short)r;
  return __builtin_bit_cast(__bf16, s);
}

// order-preserving float<->uint for atomicMax-based segment max
__device__ __forceinline__ unsigned fmapu(float x) {
  unsigned u = __float_as_uint(x);
  return (u & 0x80000000u) ? ~u : (u | 0x80000000u);
}
__device__ __forceinline__ float funmapu(unsigned u) {
  unsigned v = (u & 0x80000000u) ? (u & 0x7FFFFFFFu) : ~u;
  return __uint_as_float(v);
}

// Stage a K x 16 f32 panel (columns [n0,n0+16) of Bw[K,N]) into LDS via TDM.
// 2D descriptor per CDNA5 ISA ch.8: groups 2/3 unused (2D tensor).
__device__ __forceinline__ void tdm_load_panel(float* sB, const float* Bw,
                                               int K, int N, int n0, int lane) {
#if defined(__gfx1250__) && __has_builtin(__builtin_amdgcn_tensor_load_to_lds)
  unsigned lds_off = (unsigned)(uintptr_t)(void*)sB;      // LDS aperture: low 32 bits
  unsigned long long ga = (unsigned long long)(uintptr_t)(const void*)(Bw + (size_t)n0);
  u32x4 g0;
  g0[0] = 1u;                                             // count=1, user mode
  g0[1] = lds_off;                                        // lds_addr (bytes)
  g0[2] = (unsigned)(ga & 0xFFFFFFFFull);                 // global_addr[31:0]
  g0[3] = (unsigned)((ga >> 32) & 0x01FFFFFFull) | (2u << 30);  // addr[56:32] | type=2
  i32x8 g1;
  g1[0] = (int)(2u << 16);                                // data_size=2 (4 bytes)
  g1[1] = (int)((unsigned)N << 16);                       // tensor_dim0[15:0]
  g1[2] = (int)(((unsigned)N >> 16) | ((unsigned)K << 16));   // dim0 hi | dim1 lo
  g1[3] = (int)((((unsigned)K >> 16) & 0xFFFFu) | (16u << 16)); // dim1 hi | tile_dim0=16
  g1[4] = (int)((unsigned)K & 0xFFFFu);                   // tile_dim1=K, tile_dim2=0
  g1[5] = (int)N;                                         // tensor_dim0_stride lo
  g1[6] = 0; g1[7] = 0;
  i32x4 gz = {0, 0, 0, 0};
#if __clang_major__ >= 23
  i32x8 gp = {0, 0, 0, 0, 0, 0, 0, 0};
  __builtin_amdgcn_tensor_load_to_lds(g0, g1, gz, gz, gp, 0);
#else
  __builtin_amdgcn_tensor_load_to_lds(g0, g1, gz, gz, 0);
#endif
#if __has_builtin(__builtin_amdgcn_s_wait_tensorcnt)
  __builtin_amdgcn_s_wait_tensorcnt(0);
#endif
#else
  for (int i = lane; i < K * 16; i += 32) {
    int k = i >> 4, j = i & 15;
    sB[i] = (n0 + j < N) ? Bw[(size_t)k * N + n0 + j] : 0.f;
  }
  __syncthreads();
#endif
}

// ---------------------------------------------------------------------------
// GEMM: C[M,N] = act(A[M,K] @ B[K,N] + bias[N]); one 16x16 tile per wave32.
// Requires N%16==0 (true at all call sites); K<=128. ACT: 0=none, 3=relu.
// ---------------------------------------------------------------------------
template<int ACT>
__global__ void gemm_bias_act(const float* __restrict__ A, const float* __restrict__ Bw,
                              const float* __restrict__ bias, float* __restrict__ C,
                              int M, int K, int N) {
  __shared__ float sB[128 * 16];                 // K x 16 weight panel (K<=128)
  const int tilesN = CDIV(N, 16);
  int t  = blockIdx.x;
  int tm = t / tilesN, tn = t % tilesN;
  int l  = threadIdx.x;                          // wave32
  int lr = l & 15, half = l >> 4;
  int am = tm * 16 + lr;
  int nn = tn * 16 + lr;                         // always < N (N%16==0)
  const bool fullM = (tm * 16 + 16 <= M);
  const bool aligned = ((K & 3) == 0);

  tdm_load_panel(sB, Bw, K, N, tn * 16, l);

  v8f acc = {0.f,0.f,0.f,0.f,0.f,0.f,0.f,0.f};
  for (int k0 = 0; k0 < K; k0 += 32) {
    v16bf a, b;
    if (fullM && aligned && (k0 + 32 <= K)) {    // vector path: 4x b128 per lane
      const float* rp = A + (size_t)am * K + k0 + half * 8;
      if (k0 + 64 < K) __builtin_prefetch(rp + 64, 0, 1);   // global_prefetch_b8
      float4 c0 = *(const float4*)(rp +  0);
      float4 c1 = *(const float4*)(rp +  4);
      float4 c2 = *(const float4*)(rp + 16);
      float4 c3 = *(const float4*)(rp + 20);
      a[0]=f2bf(c0.x); a[1]=f2bf(c0.y); a[2] =f2bf(c0.z); a[3] =f2bf(c0.w);
      a[4]=f2bf(c1.x); a[5]=f2bf(c1.y); a[6] =f2bf(c1.z); a[7] =f2bf(c1.w);
      a[8]=f2bf(c2.x); a[9]=f2bf(c2.y); a[10]=f2bf(c2.z); a[11]=f2bf(c2.w);
      a[12]=f2bf(c3.x); a[13]=f2bf(c3.y); a[14]=f2bf(c3.z); a[15]=f2bf(c3.w);
    } else {                                     // edge path (K=72 convs)
#pragma unroll
      for (int j = 0; j < 16; ++j) {
        int ka = k0 + ((j & 8) ? 16 : 0) + half * 8 + (j & 7);
        a[j] = f2bf((am < M && ka < K) ? A[(size_t)am * K + ka] : 0.f);
      }
    }
#pragma unroll
    for (int j = 0; j < 16; ++j) {               // B panel from LDS (ds_load)
      int kb = k0 + ((j & 8) ? 16 : 0) + half * 8 + (j & 7);
      b[j] = f2bf((kb < K) ? sB[kb * 16 + lr] : 0.f);
    }
    acc = __builtin_amdgcn_wmma_f32_16x16x32_bf16(false, a, false, b, (short)0, acc,
                                                  false, false);
  }

  float bb = bias[nn];
  if (fullM) {                                   // branch-free epilogue
    float* cp = C + (size_t)(tm * 16 + half * 8) * N + nn;
#pragma unroll
    for (int v = 0; v < 8; ++v) {
      float r = acc[v] + bb;
      if (ACT == 3) r = fmaxf(r, 0.f);
      *cp = r; cp += N;
    }
  } else {
#pragma unroll
    for (int v = 0; v < 8; ++v) {
      int mm = tm * 16 + v + half * 8;
      if (mm < M) {
        float r = acc[v] + bb;
        if (ACT == 3) r = fmaxf(r, 0.f);
        C[(size_t)mm * N + nn] = r;
      }
    }
  }
}

// ---------------------------------------------------------------------------
// OD einsum: out[b,s,n,m] = sum_d h[s,b*N+n,d] * h[s,b*N+m,d], D=32 -> 1 WMMA.
// ---------------------------------------------------------------------------
__global__ void od_outer(const float* __restrict__ H, float* __restrict__ out,
                         int Bc, int FUTc, int N, int D) {
  const int tiles = CDIV(N, 16);
  int t  = blockIdx.x;
  int tn = t % tiles; t /= tiles;
  int tm = t % tiles; t /= tiles;
  int b  = t % Bc;
  int s  = t / Bc;
  const float* X = H + ((size_t)s * Bc * N + (size_t)b * N) * D;
  int l = threadIdx.x, lr = l & 15, half = l >> 4;
  int am = tm * 16 + lr;   // row n
  int bm = tn * 16 + lr;   // col m (B[k][m] = X[m][k])
  v16bf a, bf;
  if (am < N) {
    const float* rp = X + (size_t)am * D + half * 8;
    float4 c0 = *(const float4*)(rp +  0);
    float4 c1 = *(const float4*)(rp +  4);
    float4 c2 = *(const float4*)(rp + 16);
    float4 c3 = *(const float4*)(rp + 20);
    a[0]=f2bf(c0.x); a[1]=f2bf(c0.y); a[2] =f2bf(c0.z); a[3] =f2bf(c0.w);
    a[4]=f2bf(c1.x); a[5]=f2bf(c1.y); a[6] =f2bf(c1.z); a[7] =f2bf(c1.w);
    a[8]=f2bf(c2.x); a[9]=f2bf(c2.y); a[10]=f2bf(c2.z); a[11]=f2bf(c2.w);
    a[12]=f2bf(c3.x); a[13]=f2bf(c3.y); a[14]=f2bf(c3.z); a[15]=f2bf(c3.w);
  } else {
#pragma unroll
    for (int j = 0; j < 16; ++j) a[j] = f2bf(0.f);
  }
  if (bm < N) {                                  // two 8-float runs per lane (A layout)
    const float* rp = X + (size_t)bm * D + half * 8;
    float4 d0 = *(const float4*)(rp +  0);
    float4 d1 = *(const float4*)(rp +  4);
    float4 d2 = *(const float4*)(rp + 16);
    float4 d3 = *(const float4*)(rp + 20);
    bf[0]=f2bf(d0.x); bf[1]=f2bf(d0.y); bf[2] =f2bf(d0.z); bf[3] =f2bf(d0.w);
    bf[4]=f2bf(d1.x); bf[5]=f2bf(d1.y); bf[6] =f2bf(d1.z); bf[7] =f2bf(d1.w);
    bf[8]=f2bf(d2.x); bf[9]=f2bf(d2.y); bf[10]=f2bf(d2.z); bf[11]=f2bf(d2.w);
    bf[12]=f2bf(d3.x); bf[13]=f2bf(d3.y); bf[14]=f2bf(d3.z); bf[15]=f2bf(d3.w);
  } else {
#pragma unroll
    for (int j = 0; j < 16; ++j) bf[j] = f2bf(0.f);
  }
  v8f acc = {0.f,0.f,0.f,0.f,0.f,0.f,0.f,0.f};
  acc = __builtin_amdgcn_wmma_f32_16x16x32_bf16(false, a, false, bf, (short)0, acc,
                                                false, false);
  if (tm * 16 + 16 <= N && tn * 16 + 16 <= N) {
    float* cp = out + (((size_t)b * FUTc + s) * N + (tm * 16 + half * 8)) * N
                    + tn * 16 + lr;
#pragma unroll
    for (int v = 0; v < 8; ++v) { *cp = acc[v]; cp += N; }
  } else {
#pragma unroll
    for (int v = 0; v < 8; ++v) {
      int nr = tm * 16 + v + half * 8;
      int mc = tn * 16 + lr;
      if (nr < N && mc < N)
        out[(((size_t)b * FUTc + s) * N + nr) * N + mc] = acc[v];
    }
  }
}

// ------------------------------ GATv2 pieces -------------------------------
__global__ void gat_scores(const float* __restrict__ xl, const float* __restrict__ xr,
                           const float* __restrict__ ea, const float* __restrict__ We,
                           const float* __restrict__ att,
                           const int* __restrict__ src, const int* __restrict__ dst,
                           int E, int H, int C,
                           float* __restrict__ alpha, unsigned* __restrict__ amax) {
  int idx = blockIdx.x * blockDim.x + threadIdx.x;
  if (idx >= E * H) return;
  int e = idx / H, h = idx % H;
  int s = src[e], d = dst[e];
  float aev = (ea != nullptr) ? ea[e] : 0.f;
  const float* pl = xl + (size_t)s * H * C + h * C;
  const float* pr = xr + (size_t)d * H * C + h * C;
  const float* pw = (We != nullptr) ? (We + h * C) : nullptr;
  const float* pa = att + h * C;
  float sum = 0.f;
  for (int c = 0; c < C; ++c) {
    float m = pl[c] + pr[c] + (pw ? aev * pw[c] : 0.f);
    float g = (m > 0.f) ? m : 0.2f * m;          // leaky_relu(0.2)
    sum += g * pa[c];
  }
  alpha[idx] = sum;
  atomicMax(&amax[(size_t)d * H + h], fmapu(sum));
}

__global__ void gat_expden(float* __restrict__ alpha, const unsigned* __restrict__ amax,
                           const int* __restrict__ dst, int E, int H,
                           float* __restrict__ den) {
  int idx = blockIdx.x * blockDim.x + threadIdx.x;
  if (idx >= E * H) return;
  int e = idx / H, h = idx % H;
  float mx = funmapu(amax[(size_t)dst[e] * H + h]);
  if (!isfinite(mx)) mx = 0.f;                   // matches jnp.where(isfinite, .., 0)
  float ex = expf(alpha[idx] - mx);
  alpha[idx] = ex;
  atomicAdd(&den[(size_t)dst[e] * H + h], ex);
}

__global__ void gat_agg(const float* __restrict__ xl, const float* __restrict__ alpha,
                        const float* __restrict__ den,
                        const int* __restrict__ src, const int* __restrict__ dst,
                        int E, int H, int C, float* __restrict__ acc) {
  int idx = blockIdx.x * blockDim.x + threadIdx.x;
  if (idx >= E * H) return;
  int e = idx / H, h = idx % H;
  float w = alpha[idx] / (den[(size_t)dst[e] * H + h] + 1e-16f);
  const float* pl = xl + (size_t)src[e] * H * C + h * C;
  float* pa = acc + (size_t)dst[e] * H * C + h * C;
  for (int c = 0; c < C; ++c) atomicAdd(&pa[c], w * pl[c]);
}

__global__ void gat_fin(const float* __restrict__ acc, const float* __restrict__ bias,
                        float* __restrict__ out, int n, int H, int C,
                        int concat, int act) {
  int idx = blockIdx.x * blockDim.x + threadIdx.x;
  if (concat) {
    if (idx >= n * H * C) return;
    int hc = idx % (H * C);
    float r = acc[idx] + bias[hc];
    if (act == 1) r = (r > 0.f) ? r : expm1f(r);   // ELU
    out[idx] = r;
  } else {
    if (idx >= n * C) return;
    int i = idx / C, c = idx % C;
    float s = 0.f;
    for (int h = 0; h < H; ++h) s += acc[(size_t)i * H * C + h * C + c];
    float r = s / (float)H + bias[c];
    if (act == 1) r = (r > 0.f) ? r : expm1f(r);
    out[idx] = r;
  }
}

// ------------------------------ misc kernels -------------------------------
__global__ void build_feat(const float* __restrict__ ne, const float* __restrict__ we,
                           const int* __restrict__ wd, float* __restrict__ feat,
                           int Bc, int N, int F, int W) {
  int C = F + W;
  int idx = blockIdx.x * blockDim.x + threadIdx.x;
  if (idx >= Bc * N * C) return;
  int c = idx % C, r = idx / C;
  int n = r % N, b = r / N;
  feat[idx] = (c < F) ? ne[(size_t)n * F + c] : we[(size_t)wd[b] * W + (c - F)];
}

__global__ void vadd(const float* __restrict__ a, const float* __restrict__ b,
                     float* __restrict__ c, int n) {
  int i = blockIdx.x * blockDim.x + threadIdx.x;
  if (i < n) c[i] = a[i] + b[i];
}

__global__ void bcast_dyn(const float* __restrict__ sp, float* __restrict__ dynin,
                          int Bc, int Tc, int N, int D) {
  int idx = blockIdx.x * blockDim.x + threadIdx.x;
  if (idx >= Bc * Tc * N * D) return;
  int d = idx % D, r = idx / D;
  int n = r % N; r /= N;
  int b = r / Tc;                                 // r = b*Tc + t
  dynin[idx] = sp[((size_t)b * N + n) * D + d];
}

__global__ void build_seq(const float* __restrict__ dyn, const float* __restrict__ sp,
                          const float* __restrict__ th, const float* __restrict__ tW,
                          const float* __restrict__ tb, const float* __restrict__ posl,
                          float* __restrict__ seq, int Bc, int Tc, int N, int D) {
  int idx = blockIdx.x * blockDim.x + threadIdx.x;
  if (idx >= Bc * Tc * N * D) return;
  int d = idx % D, r = idx / D;
  int n = r % N; r /= N;
  int t = r % Tc, b = r / Tc;
  float v = dyn[idx] + sp[((size_t)b * N + n) * D + d]
          + th[(b * Tc + t) * 2 + 0] * tW[0 * D + d]
          + th[(b * Tc + t) * 2 + 1] * tW[1 * D + d] + tb[d];
  int i2 = d & ~1;
  float div = expf(-(float)i2 * 0.2878231366242557f);   // ln(10000)/32
  float ang = (float)t * div;
  v += ((d & 1) ? cosf(ang) : sinf(ang)) + posl[t * D + d];
  seq[(((size_t)t * Bc + b) * N + n) * D + d] = v;      // seq[t, b*N+n, d]
}

__global__ void mha_attn(const float* __restrict__ Q, const float* __restrict__ K,
                         const float* __restrict__ V, float* __restrict__ O,
                         int Lq, int Lk, int S, int nh, int hd) {
  int idx = blockIdx.x * blockDim.x + threadIdx.x;
  if (idx >= S * nh * Lq) return;
  int q = idx % Lq, r = idx / Lq;
  int h = r % nh,  s = r / nh;
  int D = nh * hd;
  float scale = rsqrtf((float)hd);
  const float* qp = Q + ((size_t)q * S + s) * D + h * hd;
  float sc[16];
  float mx = -1e30f;
  for (int k = 0; k < Lk; ++k) {
    const float* kp = K + ((size_t)k * S + s) * D + h * hd;
    float d = 0.f;
    for (int e = 0; e < hd; ++e) d += qp[e] * kp[e];
    d *= scale; sc[k] = d; mx = fmaxf(mx, d);
  }
  float sum = 0.f;
  for (int k = 0; k < Lk; ++k) { sc[k] = expf(sc[k] - mx); sum += sc[k]; }
  float inv = 1.f / sum;
  for (int e = 0; e < hd; ++e) {
    float o = 0.f;
    for (int k = 0; k < Lk; ++k) o += sc[k] * V[((size_t)k * S + s) * D + h * hd + e];
    O[((size_t)q * S + s) * D + h * hd + e] = o * inv;
  }
}

// X = LN(X + R); one wave32 per row (D == 32)
__global__ void layernorm_res(float* __restrict__ X, const float* __restrict__ R,
                              const float* __restrict__ g, const float* __restrict__ b,
                              int rows, int D) {
  int row  = blockIdx.x * (blockDim.x >> 5) + (threadIdx.x >> 5);
  int lane = threadIdx.x & 31;
  if (row >= rows) return;                        // uniform per wave
  bool valid = lane < D;
  float x = 0.f;
  if (valid) x = X[(size_t)row * D + lane] + R[(size_t)row * D + lane];
  float mu = x;
  for (int o = 16; o > 0; o >>= 1) mu += __shfl_xor(mu, o, 32);
  mu /= (float)D;
  float dd = valid ? (x - mu) : 0.f;
  float vv = dd * dd;
  for (int o = 16; o > 0; o >>= 1) vv += __shfl_xor(vv, o, 32);
  vv /= (float)D;
  if (valid)
    X[(size_t)row * D + lane] = dd * rsqrtf(vv + 1e-5f) * g[lane] + b[lane];
}

// ---------------------------------------------------------------------------
struct ConvP { const float *Wl,*bl,*Wr,*br,*att,*bias,*We; };
struct MhaP  { const float *wq,*bq,*wk,*bk,*wv,*bv,*wo,*bo; };
struct LnP   { const float *g,*b; };
struct DecL  { MhaP sa, ca; LnP n1,n2,n3; const float *w1,*b1,*w2,*b2; };

extern "C" void kernel_launch(void* const* d_in, const int* in_sizes, int n_in,
                              void* d_out, int out_size, void* d_ws, size_t ws_size,
                              hipStream_t stream) {
  (void)n_in; (void)out_size; (void)ws_size;
  constexpr int Bc = 4, Tc = 12, Nn = 500, FEATc = 64, WDAYc = 8, HIDc = 32,
                HEADSc = 4, HCc = 128, FUTc = 6, NHc = 8, HDh = 4, FFc = 128;
  const int BN = Bc * Nn, BTN = Bc * Tc * Nn, S = BN;
  const int Es = in_sizes[0] / 2, Ed = in_sizes[1] / 2;

  const int*   adj    = (const int*)d_in[0];
  const int*   asrc   = adj,          *adst = adj + Es;
  const int*   dei    = (const int*)d_in[1];
  const int*   dsrc   = dei,          *ddst = dei + Ed;
  const float* dea    = (const float*)d_in[2];
  const float* thist  = (const float*)d_in[3];
  const int*   wkday  = (const int*)d_in[4];
  // d_in[5]=B, d_in[6]=T (device scalars; compile-time constants used instead)

  // ---- params: jax pytree order (dict keys sorted, lists in order) --------
  auto F = [&](int i) { return (const float*)d_in[i]; };
  int ip = 7;
  DecL dl[2];
  for (int L = 0; L < 2; ++L) {           // dec
    DecL& d = dl[L];
    d.b1 = F(ip++); d.b2 = F(ip++);
    d.ca.bk = F(ip++); d.ca.bo = F(ip++); d.ca.bq = F(ip++); d.ca.bv = F(ip++);
    d.ca.wk = F(ip++); d.ca.wo = F(ip++); d.ca.wq = F(ip++); d.ca.wv = F(ip++);
    d.n1.b = F(ip++); d.n1.g = F(ip++);
    d.n2.b = F(ip++); d.n2.g = F(ip++);
    d.n3.b = F(ip++); d.n3.g = F(ip++);
    d.sa.bk = F(ip++); d.sa.bo = F(ip++); d.sa.bq = F(ip++); d.sa.bv = F(ip++);
    d.sa.wk = F(ip++); d.sa.wo = F(ip++); d.sa.wq = F(ip++); d.sa.wv = F(ip++);
    d.w1 = F(ip++); d.w2 = F(ip++);
  }
  ConvP dynp[6];
  for (int e = 0; e < 3; ++e)             // dyn (conv has We; keys sorted)
    for (int c = 0; c < 2; ++c) {
      ConvP& p = dynp[e * 2 + c];
      p.We = F(ip++); p.Wl = F(ip++); p.Wr = F(ip++);
      p.att = F(ip++); p.bias = F(ip++); p.bl = F(ip++); p.br = F(ip++);
    }
  ConvP longp[8];
  for (int e = 0; e < 4; ++e)             // long
    for (int c = 0; c < 2; ++c) {
      ConvP& p = longp[e * 2 + c];
      p.We = nullptr;
      p.Wl = F(ip++); p.Wr = F(ip++); p.att = F(ip++);
      p.bias = F(ip++); p.bl = F(ip++); p.br = F(ip++);
    }
  const float* node_embed = F(ip++);      // node_embed
  const float* pos_learn  = F(ip++);      // pos_learn
  ConvP shortp[4];
  for (int e = 0; e < 2; ++e)             // short
    for (int c = 0; c < 2; ++c) {
      ConvP& p = shortp[e * 2 + c];
      p.We = nullptr;
      p.Wl = F(ip++); p.Wr = F(ip++); p.att = F(ip++);
      p.bias = F(ip++); p.bl = F(ip++); p.br = F(ip++);
    }
  const float* time_W = F(ip++);
  const float* time_b = F(ip++);
  const float* wk_emb = F(ip++);

  // ---- workspace bump allocator -------------------------------------------
  char* wsb = (char*)d_ws; size_t off = 0;
  auto alloc = [&](size_t nfl) {
    void* p = wsb + off; off += ((nfl + 3) & ~size_t(3)) * sizeof(float); return p;
  };
  float*    featb   = (float*)alloc((size_t)BN * 72);
  float*    bufA    = (float*)alloc((size_t)BTN * HCc);
  float*    bufB    = (float*)alloc((size_t)BTN * HCc);
  float*    xlb     = (float*)alloc((size_t)BTN * HCc);
  float*    xrb     = (float*)alloc((size_t)BTN * HCc);
  float*    accb    = (float*)alloc((size_t)BTN * HCc);
  float*    alphab  = (float*)alloc((size_t)Ed * HEADSc);
  unsigned* amaxb   = (unsigned*)alloc((size_t)BTN * HEADSc);
  float*    denb    = (float*)alloc((size_t)BTN * HEADSc);
  float*    shortO  = (float*)alloc((size_t)BN * HIDc);
  float*    longO   = (float*)alloc((size_t)BN * HIDc);
  float*    spat    = (float*)alloc((size_t)BN * HIDc);
  float*    dyninb  = (float*)alloc((size_t)BTN * HIDc);
  float*    dynOb   = (float*)alloc((size_t)BTN * HIDc);
  float*    seqb    = (float*)alloc((size_t)Tc * S * HIDc);
  float*    hdec    = (float*)alloc((size_t)FUTc * S * HIDc);
  float*    Qb      = (float*)alloc((size_t)Tc * S * HIDc);
  float*    Kb      = (float*)alloc((size_t)Tc * S * HIDc);
  float*    Vb      = (float*)alloc((size_t)Tc * S * HIDc);
  float*    attO    = (float*)alloc((size_t)FUTc * S * HIDc);
  float*    projb   = (float*)alloc((size_t)FUTc * S * HIDc);
  float*    ffn1    = (float*)alloc((size_t)FUTc * S * FFc);

  auto gemm0 = [&](const float* A, const float* W, const float* bias, float* C,
                   int M, int K, int N) {
    gemm_bias_act<0><<<CDIV(M, 16) * CDIV(N, 16), 32, 0, stream>>>(A, W, bias, C, M, K, N);
  };

  auto run_conv = [&](const float* x, int n, int cin, int H, int C, int concat,
                      const int* src, const int* dst, int E, const float* ea,
                      const ConvP& p, float* out, int act) {
    int HCo = H * C;
    gemm0(x, p.Wl, p.bl, xlb, n, cin, HCo);
    gemm0(x, p.Wr, p.br, xrb, n, cin, HCo);
    (void)hipMemsetAsync(amaxb, 0, (size_t)n * H * sizeof(unsigned), stream);
    (void)hipMemsetAsync(denb,  0, (size_t)n * H * sizeof(float),    stream);
    (void)hipMemsetAsync(accb,  0, (size_t)n * HCo * sizeof(float),  stream);
    int EH = E * H;
    gat_scores<<<CDIV(EH, 256), 256, 0, stream>>>(xlb, xrb, ea, p.We, p.att,
                                                  src, dst, E, H, C, alphab, amaxb);
    gat_expden<<<CDIV(EH, 256), 256, 0, stream>>>(alphab, amaxb, dst, E, H, denb);
    gat_agg<<<CDIV(EH, 256), 256, 0, stream>>>(xlb, alphab, denb, src, dst, E, H, C, accb);
    int od = concat ? HCo : C;
    gat_fin<<<CDIV(n * od, 256), 256, 0, stream>>>(accb, p.bias, out, n, H, C, concat, act);
  };

  auto run_stack = [&](const float* x0, int n, int cin0, const ConvP* cp, int nenc,
                       const int* src, const int* dst, int E, const float* ea,
                       float* outb) {
    const float* cur = x0; int cin = cin0;
    for (int e = 0; e < nenc; ++e) {
      run_conv(cur, n, cin, HEADSc, HIDc, 1, src, dst, E, ea, cp[e * 2 + 0], bufA, 1);
      float* o2 = (e == nenc - 1) ? outb : bufB;
      run_conv(bufA, n, HEADSc * HIDc, 1, HIDc, 0, src, dst, E, ea, cp[e * 2 + 1], o2, 1);
      cur = o2; cin = HIDc;
    }
  };

  // ---- spatial encoders ---------------------------------------------------
  build_feat<<<CDIV(BN * (FEATc + WDAYc), 256), 256, 0, stream>>>(
      node_embed, wk_emb, wkday, featb, Bc, Nn, FEATc, WDAYc);
  run_stack(featb, BN, FEATc + WDAYc, shortp, 2, asrc, adst, Es, nullptr, shortO);
  run_stack(featb, BN, FEATc + WDAYc, longp,  4, asrc, adst, Es, nullptr, longO);
  vadd<<<CDIV(BN * HIDc, 256), 256, 0, stream>>>(shortO, longO, spat, BN * HIDc);

  // ---- dynamic encoders ---------------------------------------------------
  bcast_dyn<<<CDIV(BTN * HIDc, 256), 256, 0, stream>>>(spat, dyninb, Bc, Tc, Nn, HIDc);
  run_stack(dyninb, BTN, HIDc, dynp, 3, dsrc, ddst, Ed, dea, dynOb);

  // ---- fuse + positional encodings ---------------------------------------
  build_seq<<<CDIV(BTN * HIDc, 256), 256, 0, stream>>>(
      dynOb, spat, thist, time_W, time_b, pos_learn, seqb, Bc, Tc, Nn, HIDc);

  // ---- transformer decoder (post-LN), tgt = zeros -------------------------
  (void)hipMemsetAsync(hdec, 0, (size_t)FUTc * S * HIDc * sizeof(float), stream);
  for (int L = 0; L < 2; ++L) {
    const DecL& d = dl[L];
    // self-attention
    gemm0(hdec, d.sa.wq, d.sa.bq, Qb, FUTc * S, HIDc, HIDc);
    gemm0(hdec, d.sa.wk, d.sa.bk, Kb, FUTc * S, HIDc, HIDc);
    gemm0(hdec, d.sa.wv, d.sa.bv, Vb, FUTc * S, HIDc, HIDc);
    mha_attn<<<CDIV(S * NHc * FUTc, 256), 256, 0, stream>>>(Qb, Kb, Vb, attO,
                                                            FUTc, FUTc, S, NHc, HDh);
    gemm0(attO, d.sa.wo, d.sa.bo, projb, FUTc * S, HIDc, HIDc);
    layernorm_res<<<CDIV(FUTc * S, 8), 256, 0, stream>>>(hdec, projb, d.n1.g, d.n1.b,
                                                         FUTc * S, HIDc);
    // cross-attention to seq
    gemm0(hdec, d.ca.wq, d.ca.bq, Qb, FUTc * S, HIDc, HIDc);
    gemm0(seqb, d.ca.wk, d.ca.bk, Kb, Tc * S, HIDc, HIDc);
    gemm0(seqb, d.ca.wv, d.ca.bv, Vb, Tc * S, HIDc, HIDc);
    mha_attn<<<CDIV(S * NHc * FUTc, 256), 256, 0, stream>>>(Qb, Kb, Vb, attO,
                                                            FUTc, Tc, S, NHc, HDh);
    gemm0(attO, d.ca.wo, d.ca.bo, projb, FUTc * S, HIDc, HIDc);
    layernorm_res<<<CDIV(FUTc * S, 8), 256, 0, stream>>>(hdec, projb, d.n2.g, d.n2.b,
                                                         FUTc * S, HIDc);
    // FFN (relu)
    gemm_bias_act<3><<<CDIV(FUTc * S, 16) * CDIV(FFc, 16), 32, 0, stream>>>(
        hdec, d.w1, d.b1, ffn1, FUTc * S, HIDc, FFc);
    gemm0(ffn1, d.w2, d.b2, projb, FUTc * S, FFc, HIDc);
    layernorm_res<<<CDIV(FUTc * S, 8), 256, 0, stream>>>(hdec, projb, d.n3.g, d.n3.b,
                                                         FUTc * S, HIDc);
  }

  // ---- OD matrices: out[b,s] = dec dec^T (K=32 -> one WMMA per tile) ------
  int tiles = CDIV(Nn, 16);
  od_outer<<<FUTc * Bc * tiles * tiles, 32, 0, stream>>>(hdec, (float*)d_out,
                                                         Bc, FUTc, Nn, HIDc);
}
